// BitVocoder_35605278884133
// MI455X (gfx1250) — compile-verified, matrix operand order checked by reading, not run on hardware
//
#include <hip/hip_runtime.h>
#include <hip/hip_bf16.h>
#include <math.h>

#define DIMC 256
#define TT   2048
#define BATCH 8
#define MM   (BATCH*TT)          /* 16384 rows */
#define NFFT 1024
#define HOP  320
#define FB   513
#define KPAD 1056                /* 2*FB padded to multiple of 32 */
#define LOUT  (HOP*(TT-1))       /* 655040 */
#define PI_F 3.14159265358979323846f
#define TWO_PI_F 6.28318530717958647692f

typedef __attribute__((ext_vector_type(16))) __bf16 v16bf;
typedef __attribute__((ext_vector_type(8)))  float  v8f;

union FragBF { uint4 u[2]; v16bf v; };

__device__ __forceinline__ unsigned short f2bf(float f) {
  unsigned u = __float_as_uint(f);
  unsigned r = u + 0x7FFFu + ((u >> 16) & 1u);   // round-to-nearest-even
  return (unsigned short)(r >> 16);
}

// ---------------------------------------------------------------- reductions
__global__ __launch_bounds__(256) void k_absmean(const float* __restrict__ w, int n,
                                                 float* __restrict__ out) {
  __shared__ float sm[256];
  float s = 0.f;
  for (int i = threadIdx.x; i < n; i += 256) s += fabsf(w[i]);
  sm[threadIdx.x] = s;
  __syncthreads();
  for (int st = 128; st > 0; st >>= 1) {
    if ((int)threadIdx.x < st) sm[threadIdx.x] += sm[threadIdx.x + st];
    __syncthreads();
  }
  if (threadIdx.x == 0) out[0] = fmaxf(sm[0] / (float)n, 1e-5f);
}

// ternary values {-1,0,1} are exact in bf16; scale folded into GEMM epilogue
__global__ void k_ternarize(const float* __restrict__ w, const float* __restrict__ s,
                            unsigned short* __restrict__ q, int n) {
  int i = blockIdx.x * 256 + threadIdx.x;
  if (i >= n) return;
  float v = rintf(w[i] / s[0]);
  v = fminf(fmaxf(v, -1.f), 1.f);
  q[i] = f2bf(v);
}

// (O,C,taps) -> tap-major (taps,O,C) so each tap slice is a contiguous (N,K) matrix
__global__ void k_ternarize_conv(const float* __restrict__ w, const float* __restrict__ s,
                                 unsigned short* __restrict__ q, int N, int C, int taps) {
  int n = N * C * taps;
  int i = blockIdx.x * 256 + threadIdx.x;
  if (i >= n) return;
  int tap = i / (N * C);
  int rem = i - tap * N * C;
  int o = rem / C, c = rem - o * C;
  float v = rintf(w[(o * C + c) * taps + tap] / s[0]);
  v = fminf(fmaxf(v, -1.f), 1.f);
  q[i] = f2bf(v);
}

// ---------------------------------------------------------------- layout / elementwise
__global__ void k_x_to_btc_bf16(const float* __restrict__ x, unsigned short* __restrict__ o) {
  int i = blockIdx.x * 256 + threadIdx.x;
  if (i >= MM * DIMC) return;
  int c = i & (DIMC - 1);
  int t = (i >> 8) & (TT - 1);
  int b = i >> 19;
  o[i] = f2bf(x[((size_t)(b * DIMC + c)) * TT + t]);
}

__global__ void k_dwconv7(const float* __restrict__ x, const float* __restrict__ w,
                          const float* __restrict__ bias, float* __restrict__ y, int dil) {
  int i = blockIdx.x * 256 + threadIdx.x;
  if (i >= MM * DIMC) return;
  int c = i & (DIMC - 1);
  int m = i >> 8;
  int t = m & (TT - 1);
  int b0 = m - t;            // b*TT
  float acc = bias[c];
#pragma unroll
  for (int j = 0; j < 7; ++j) {
    int ts = t + dil * (j - 3);
    if ((unsigned)ts < (unsigned)TT)
      acc += w[c * 7 + j] * x[(((size_t)(b0 + ts)) << 8) + c];
  }
  y[i] = acc;
}

__global__ __launch_bounds__(256) void k_rmsnorm(const float* __restrict__ x,
                                                 const float* __restrict__ w,
                                                 unsigned short* __restrict__ o) {
  int row = blockIdx.x * 8 + (threadIdx.x >> 5);     // wave32 per row
  int lane = threadIdx.x & 31;
  const float* xr = x + (size_t)row * DIMC;
  float v[8]; float ss = 0.f;
#pragma unroll
  for (int j = 0; j < 8; ++j) { float f = xr[lane + 32 * j]; v[j] = f; ss += f * f; }
#pragma unroll
  for (int off = 16; off > 0; off >>= 1) ss += __shfl_xor(ss, off, 32);
  float r = rsqrtf(ss * (1.f / DIMC) + 1e-6f);
  unsigned short* orow = o + (size_t)row * DIMC;
#pragma unroll
  for (int j = 0; j < 8; ++j) { int c = lane + 32 * j; orow[c] = f2bf(v[j] * r * w[c]); }
}

__global__ void k_snake(const float* __restrict__ x, const float* __restrict__ a,
                        const float* __restrict__ be, float* yf, unsigned short* ybf,
                        int C, int n) {
  int i = blockIdx.x * 256 + threadIdx.x;
  if (i >= n) return;
  int c = i & (C - 1);       // C is a power of two (256 or 1024)
  float v = x[i];
  float sn = sinf(a[c] * v);
  float r = v + sn * sn / (be[c] + 1e-9f);
  if (yf)  yf[i] = r;
  if (ybf) ybf[i] = f2bf(r);
}

__global__ void k_add(float* __restrict__ y, const float* __restrict__ x, int n) {
  int i = blockIdx.x * 256 + threadIdx.x;
  if (i < n) y[i] += x[i];
}

__global__ void k_tobf16(const float* __restrict__ x, unsigned short* __restrict__ o, int n) {
  int i = blockIdx.x * 256 + threadIdx.x;
  if (i < n) o[i] = f2bf(x[i]);
}

// ---------------------------------------------------------------- WMMA GEMM
// Y[M,N] = epilogue( sum_tap A_shift(tap)[M,K] * W[tap][N,K]^T )
// A: bf16 (B,T,K) row-major; tap shifts rows in time with zero padding.
// Block tile 64(M) x 128(N), BK=32, 256 threads = 8 wave32 (4M x 2N),
// each wave: 1 A-frag x 4 B-frags = 4 v_wmma per K-step, issued back-to-back
// after all fragments are in registers (single s_wait_dscnt per step).
// LDS double-buffered: one barrier per K-step, global loads overlap WMMA.
// OOB handled by clamped addresses + data masking (no divergent branches).
__global__ __launch_bounds__(256) void k_gemm(
    const unsigned short* __restrict__ A, const unsigned short* __restrict__ W,
    const float* __restrict__ scale, const float* __restrict__ bias,
    const float* __restrict__ gamma, const float* resid, float* Y,
    int N, int K, int taps, int dil, int center) {
  __shared__ __align__(16) unsigned short As[2][64 * 32];
  __shared__ __align__(16) unsigned short Bs[2][128 * 32];
  const int m0 = blockIdx.x * 64;
  const int n0 = blockIdx.y * 128;
  const int tid = threadIdx.x;
  const int lane = tid & 31;
  const int wave = tid >> 5;
  const int wm = wave & 3;
  const int wn = wave >> 2;

  // cooperative-load coordinates: thread covers 8 contiguous bf16 (16B) per tile row
  const int arow = tid >> 2;          // 0..63
  const int lcg = (tid & 3) << 3;     // 0,8,16,24
  const int mA = m0 + arow;
  const int tIdx = mA & (TT - 1);
  const int bBase = mA - tIdx;        // b*TT
  const int nB1 = n0 + arow;
  const int nB2 = nB1 + 64;
  const bool bval1 = (nB1 < N);
  const bool bval2 = (nB2 < N);
  const size_t bOff1 = (size_t)(bval1 ? nB1 : 0) * (size_t)K;
  const size_t bOff2 = (size_t)(bval2 ? nB2 : 0) * (size_t)K;
  const size_t wTap = (size_t)N * (size_t)K;

  const int ksteps = K >> 5;
  const int L = taps * ksteps;

  int tapc = 0, kc = 0;               // uniform counters, advanced incrementally
  const uint4 z4 = {0u, 0u, 0u, 0u};

  auto ldg = [&](uint4& av, uint4& b1, uint4& b2) {
    int ts = tIdx + dil * (tapc - center);
    bool aval = (unsigned)ts < (unsigned)TT;
    int tsc = aval ? ts : 0;          // clamped -> always a valid address
    av = *(const uint4*)(A + (size_t)(bBase + tsc) * (size_t)K + (kc + lcg));
    if (!aval) av = z4;
    const unsigned short* wb = W + (size_t)tapc * wTap + (kc + lcg);
    b1 = *(const uint4*)(wb + bOff1); if (!bval1) b1 = z4;
    b2 = *(const uint4*)(wb + bOff2); if (!bval2) b2 = z4;
    kc += 32; if (kc == K) { kc = 0; ++tapc; }
  };
  auto sts = [&](int pp, const uint4& av, const uint4& b1, const uint4& b2) {
    *(uint4*)&As[pp][arow * 32 + lcg] = av;
    *(uint4*)&Bs[pp][arow * 32 + lcg] = b1;
    *(uint4*)&Bs[pp][(arow + 64) * 32 + lcg] = b2;
  };

  // fragment addresses (loop-invariant); bf16 16x16x32 K-map:
  // halves 0-7 -> K=kb..kb+7, halves 8-15 -> K=kb+16..kb+23, kb = 8*lane[4]
  const int fr = lane & 15;
  const int kb = (lane >> 4) << 3;
  const int aoff = (wm * 16 + fr) * 32 + kb;
  const int boff = (wn * 64 + fr) * 32 + kb;

  v8f acc[4];
#pragma unroll
  for (int j = 0; j < 4; ++j) acc[j] = (v8f){0.f,0.f,0.f,0.f,0.f,0.f,0.f,0.f};

  uint4 av, b1, b2;
  ldg(av, b1, b2);
  sts(0, av, b1, b2);
  int p = 0;
  for (int i = 0; i < L; ++i) {
    __syncthreads();                  // buf[p] ready; prior reads of buf[p^1] done
    const bool more = (i + 1) < L;
    if (more) ldg(av, b1, b2);        // overlap next global tile with WMMAs
    // preload ALL fragments, then issue 4 WMMAs back-to-back
    FragBF fa, fb[4];
    fa.u[0] = *(const uint4*)&As[p][aoff];
    fa.u[1] = *(const uint4*)&As[p][aoff + 16];
#pragma unroll
    for (int j = 0; j < 4; ++j) {
      const unsigned short* bp = &Bs[p][boff + j * (16 * 32)];
      fb[j].u[0] = *(const uint4*)bp;
      fb[j].u[1] = *(const uint4*)(bp + 16);
    }
#pragma unroll
    for (int j = 0; j < 4; ++j)
      acc[j] = __builtin_amdgcn_wmma_f32_16x16x32_bf16(false, fa.v, false, fb[j].v,
                                                       (short)0, acc[j], false, false);
    if (more) sts(p ^ 1, av, b1, b2);
    p ^= 1;
  }

  // epilogue: Y = resid + gamma * (acc*scale + bias)
  const float s = scale ? scale[0] : 1.f;
  const int mrow = m0 + wm * 16 + ((lane >> 4) << 3);
  int nc[4]; float bb[4], gg[4];
#pragma unroll
  for (int j = 0; j < 4; ++j) {
    nc[j] = n0 + wn * 64 + 16 * j + fr;
    bb[j] = (bias  && nc[j] < N) ? bias[nc[j]]  : 0.f;
    gg[j] = (gamma && nc[j] < N) ? gamma[nc[j]] : 1.f;
  }
#pragma unroll
  for (int r = 0; r < 8; ++r) {
    const size_t ro = (size_t)(mrow + r) * (size_t)N;
#pragma unroll
    for (int j = 0; j < 4; ++j) {
      if (nc[j] < N) {
        float v = (acc[j][r] * s + bb[j]) * gg[j];
        if (resid) v += resid[ro + nc[j]];
        Y[ro + nc[j]] = v;
      }
    }
  }
}

// ---------------------------------------------------------------- heads / iSTFT
__global__ void k_tanhpi(float* p, const float* __restrict__ cw, int n) {
  int i = blockIdx.x * 256 + threadIdx.x;
  if (i < n) p[i] = tanhf(p[i]) * PI_F * cw[0];
}

__global__ void k_cumsum(float* p) {            // thread per (b,f) sequence
  int i = blockIdx.x * 256 + threadIdx.x;
  if (i >= BATCH * FB) return;
  int b = i / FB, f = i - b * FB;
  size_t base = (size_t)b * TT * FB + f;
  float acc = 0.f;
  for (int t = 0; t < TT; ++t) {
    size_t o = base + (size_t)t * FB;
    acc += p[o];
    float q = acc + PI_F;
    q -= TWO_PI_F * floorf(q * (1.f / TWO_PI_F));
    p[o] = q - PI_F;                            // wrapped phase in [-pi,pi)
  }
}

// pack [Re(513) | Im(513) | 0-pad] per frame as bf16 GEMM-A rows of width KPAD
__global__ void k_specpack(const float* __restrict__ lm, const float* __restrict__ ph,
                           unsigned short* __restrict__ sp) {
  int i = blockIdx.x * 256 + threadIdx.x;
  if (i >= MM * KPAD) return;
  int k = i % KPAD;
  int m = i / KPAD;
  unsigned short v = 0;
  if (k < 2 * FB) {
    int f = (k < FB) ? k : (k - FB);
    size_t o = (size_t)m * FB + f;
    float l = fminf(fmaxf(lm[o], -10.f), 10.f);
    float mg = expf(l);
    float phase = ph[o];
    v = f2bf(mg * ((k < FB) ? cosf(phase) : sinf(phase)));
  }
  sp[i] = v;
}

// irfft x hann as a (1024, KPAD) bf16 basis: y[n] = sum_k Br[k,n]Re_k + Bi[k,n]Im_k
__global__ void k_basis(unsigned short* __restrict__ bs) {
  int i = blockIdx.x * 256 + threadIdx.x;
  if (i >= NFFT * KPAD) return;
  int k = i % KPAD;
  int n = i / KPAD;
  float win = 0.5f - 0.5f * cosf(TWO_PI_F * (float)n * (1.f / NFFT));
  float val = 0.f;
  if (k < FB) {
    float c = (k == 0 || k == FB - 1) ? 1.f : 2.f;
    int phi = (k * n) & (NFFT - 1);             // exact angle reduction
    val = c * cosf(TWO_PI_F * (float)phi * (1.f / NFFT)) * (1.f / NFFT) * win;
  } else if (k < 2 * FB) {
    int kk = k - FB;
    if (kk != 0 && kk != FB - 1) {              // Im of DC/Nyquist ignored by irfft
      int phi = (kk * n) & (NFFT - 1);
      val = -2.f * sinf(TWO_PI_F * (float)phi * (1.f / NFFT)) * (1.f / NFFT) * win;
    }
  }
  bs[i] = f2bf(val);
}

// deterministic overlap-add gather + envelope divide + center trim
__global__ void k_istft_out(const float* __restrict__ fr, float* __restrict__ out) {
  int i = blockIdx.x * 256 + threadIdx.x;
  if (i >= BATCH * LOUT) return;
  int b = i / LOUT;
  int l = (i - b * LOUT) + NFFT / 2;
  int fmin = (l - (NFFT - 1) + (HOP - 1)) / HOP; if (fmin < 0) fmin = 0;
  int fmax = l / HOP;                            if (fmax > TT - 1) fmax = TT - 1;
  float acc = 0.f, env = 0.f;
  for (int f = fmin; f <= fmax; ++f) {
    int n = l - f * HOP;
    acc += fr[((size_t)(b * TT + f) << 10) + n];
    float w = 0.5f - 0.5f * cosf(TWO_PI_F * (float)n * (1.f / NFFT));
    env += w * w;
  }
  out[i] = acc / ((env > 1e-11f) ? env : 1.f);
}

// ---------------------------------------------------------------- host side
struct Params {
  const float *x;
  const float *conv_in_w, *conv_in_b, *act_in_a, *act_in_be, *skip_w, *skip_b;
  const float *dw_w[4], *dw_b[4], *bnorm_w[4], *pw1_w[4], *pw1_b[4],
              *a1[4], *be1[4], *pw2_w[4], *pw2_b[4], *gamma[4];
  const float *rw1[3], *rb1[3], *ra[3], *rbe[3], *rw2[3], *rb2[3];
  const float *norm_w, *mag1_w, *mag1_b, *mag_a, *mag_be,
              *mag2_w, *mag2_b, *ph_w, *ph_b, *cumsum_w;
};

static void map_params(void* const* d_in, const int* in_sizes, int n_in, Params& p) {
  (void)n_in;
  int i = 0;
  auto F = [&](void) { return (const float*)d_in[i++]; };
  if (in_sizes[0] == BATCH * DIMC * TT) {
    // dict insertion order: x first, params flattened in insertion order
    p.x = F();
    p.conv_in_w = F(); p.conv_in_b = F(); p.act_in_a = F(); p.act_in_be = F();
    p.skip_w = F(); p.skip_b = F();
    for (int k = 0; k < 4; ++k) {
      p.dw_w[k] = F(); p.dw_b[k] = F(); p.bnorm_w[k] = F();
      p.pw1_w[k] = F(); p.pw1_b[k] = F(); p.a1[k] = F(); p.be1[k] = F();
      p.pw2_w[k] = F(); p.pw2_b[k] = F(); p.gamma[k] = F();
    }
    for (int j = 0; j < 3; ++j) {
      p.rw1[j] = F(); p.rb1[j] = F(); p.ra[j] = F(); p.rbe[j] = F();
      p.rw2[j] = F(); p.rb2[j] = F();
    }
    p.norm_w = F(); p.mag1_w = F(); p.mag1_b = F(); p.mag_a = F(); p.mag_be = F();
    p.mag2_w = F(); p.mag2_b = F(); p.ph_w = F(); p.ph_b = F(); p.cumsum_w = F();
  } else {
    // jax tree_leaves order: dict keys sorted at every level, 'params' < 'x'
    p.act_in_a = F(); p.act_in_be = F();
    for (int k = 0; k < 4; ++k) {
      p.a1[k] = F(); p.be1[k] = F(); p.dw_b[k] = F(); p.dw_w[k] = F();
      p.gamma[k] = F(); p.bnorm_w[k] = F();
      p.pw1_b[k] = F(); p.pw1_w[k] = F(); p.pw2_b[k] = F(); p.pw2_w[k] = F();
    }
    p.conv_in_b = F(); p.conv_in_w = F(); p.cumsum_w = F();
    p.mag1_b = F(); p.mag1_w = F(); p.mag2_b = F(); p.mag2_w = F();
    p.mag_a = F(); p.mag_be = F(); p.norm_w = F(); p.ph_b = F(); p.ph_w = F();
    for (int j = 0; j < 3; ++j) {
      p.ra[j] = F(); p.rb1[j] = F(); p.rb2[j] = F(); p.rbe[j] = F();
      p.rw1[j] = F(); p.rw2[j] = F();
    }
    p.skip_b = F(); p.skip_w = F(); p.x = F();
  }
}

extern "C" void kernel_launch(void* const* d_in, const int* in_sizes, int n_in,
                              void* d_out, int out_size, void* d_ws, size_t ws_size,
                              hipStream_t stream) {
  (void)out_size; (void)ws_size;
  Params p;
  map_params(d_in, in_sizes, n_in, p);

  char* base = (char*)d_ws;
  size_t off = 0;
  auto carve = [&](size_t bytes) -> void* {
    off = (off + 255) & ~(size_t)255;
    void* ptr = base + off;
    off += bytes;
    return ptr;
  };
  float* hbuf  = (float*)carve((size_t)MM * DIMC * 4);
  float* skipb = (float*)carve((size_t)MM * DIMC * 4);
  float* t1    = (float*)carve((size_t)MM * DIMC * 4);
  float* t4    = (float*)carve((size_t)MM * 1024 * 4);     // also iSTFT frames
  unsigned short* a256  = (unsigned short*)carve((size_t)MM * DIMC * 2);
  unsigned short* a1024 = (unsigned short*)carve((size_t)MM * 1024 * 2);
  float* lm2 = (float*)carve((size_t)MM * FB * 4);
  float* phb = (float*)carve((size_t)MM * FB * 4);
  unsigned short* spack  = (unsigned short*)carve((size_t)MM * KPAD * 2);
  unsigned short* basisb = (unsigned short*)carve((size_t)NFFT * KPAD * 2);
  float* scales = (float*)carve(32 * 4);
  unsigned short* q_skip = (unsigned short*)carve((size_t)DIMC * DIMC * 2);
  unsigned short* q_cin  = (unsigned short*)carve((size_t)7 * DIMC * DIMC * 2);
  unsigned short *q_pw1[4], *q_pw2[4];
  for (int k = 0; k < 4; ++k) {
    q_pw1[k] = (unsigned short*)carve((size_t)1024 * DIMC * 2);
    q_pw2[k] = (unsigned short*)carve((size_t)DIMC * 1024 * 2);
  }
  unsigned short *q_rw1[3], *q_rw2[3];
  for (int j = 0; j < 3; ++j) {
    q_rw1[j] = (unsigned short*)carve((size_t)3 * DIMC * DIMC * 2);
    q_rw2[j] = (unsigned short*)carve((size_t)3 * DIMC * DIMC * 2);
  }
  unsigned short* q_mag1 = (unsigned short*)carve((size_t)DIMC * DIMC * 2);
  unsigned short* q_mag2 = (unsigned short*)carve((size_t)FB * DIMC * 2);
  unsigned short* q_ph   = (unsigned short*)carve((size_t)FB * DIMC * 2);

  auto qplain = [&](const float* w, int n, int sidx, unsigned short* dst) {
    k_absmean<<<1, 256, 0, stream>>>(w, n, scales + sidx);
    k_ternarize<<<(n + 255) / 256, 256, 0, stream>>>(w, scales + sidx, dst, n);
  };
  auto qconv = [&](const float* w, int N, int C, int taps, int sidx, unsigned short* dst) {
    int n = N * C * taps;
    k_absmean<<<1, 256, 0, stream>>>(w, n, scales + sidx);
    k_ternarize_conv<<<(n + 255) / 256, 256, 0, stream>>>(w, scales + sidx, dst, N, C, taps);
  };
  auto gemm = [&](const unsigned short* A, const unsigned short* W, const float* sc,
                  const float* bias, const float* gm, const float* rs, float* Y,
                  int N, int K, int taps, int dil, int center) {
    dim3 grid(MM / 64, (N + 127) / 128);
    k_gemm<<<grid, 256, 0, stream>>>(A, W, sc, bias, gm, rs, Y, N, K, taps, dil, center);
  };

  // iDFT basis (independent of inputs)
  k_basis<<<(NFFT * KPAD + 255) / 256, 256, 0, stream>>>(basisb);

  // x: (B,C,T) -> bf16 (B,T,C)
  k_x_to_btc_bf16<<<(MM * DIMC + 255) / 256, 256, 0, stream>>>(p.x, a256);

  // absmean ternarization of every quantized weight (bf16-exact {-1,0,1})
  qplain(p.skip_w, DIMC * DIMC, 0, q_skip);
  qconv(p.conv_in_w, DIMC, DIMC, 7, 1, q_cin);
  for (int k = 0; k < 4; ++k) {
    qplain(p.pw1_w[k], 1024 * DIMC, 2 + k, q_pw1[k]);
    qplain(p.pw2_w[k], DIMC * 1024, 6 + k, q_pw2[k]);
  }
  for (int j = 0; j < 3; ++j) {
    qconv(p.rw1[j], DIMC, DIMC, 3, 10 + j, q_rw1[j]);
    qconv(p.rw2[j], DIMC, DIMC, 3, 13 + j, q_rw2[j]);
  }
  qplain(p.mag1_w, DIMC * DIMC, 16, q_mag1);
  qplain(p.mag2_w, FB * DIMC, 17, q_mag2);
  qplain(p.ph_w,   FB * DIMC, 18, q_ph);

  // skip (1x1) and conv_in (k=7) as WMMA GEMMs
  gemm(a256, q_skip, scales + 0, p.skip_b, nullptr, nullptr, skipb, DIMC, DIMC, 1, 0, 0);
  gemm(a256, q_cin,  scales + 1, p.conv_in_b, nullptr, nullptr, t1, DIMC, DIMC, 7, 1, 3);
  k_snake<<<(MM * DIMC + 255) / 256, 256, 0, stream>>>(t1, p.act_in_a, p.act_in_be,
                                                       hbuf, (unsigned short*)nullptr,
                                                       DIMC, MM * DIMC);

  // ConvNeXt blocks, dilations 1,2,4,8
  const int dils[4] = {1, 2, 4, 8};
  for (int k = 0; k < 4; ++k) {
    k_dwconv7<<<(MM * DIMC + 255) / 256, 256, 0, stream>>>(hbuf, p.dw_w[k], p.dw_b[k],
                                                           t1, dils[k]);
    k_rmsnorm<<<MM / 8, 256, 0, stream>>>(t1, p.bnorm_w[k], a256);
    gemm(a256, q_pw1[k], scales + 2 + k, p.pw1_b[k], nullptr, nullptr, t4, 1024, DIMC, 1, 0, 0);
    k_snake<<<(MM * 1024 + 255) / 256, 256, 0, stream>>>(t4, p.a1[k], p.be1[k],
                                                         (float*)nullptr, a1024,
                                                         1024, MM * 1024);
    gemm(a1024, q_pw2[k], scales + 6 + k, p.pw2_b[k], p.gamma[k], hbuf, hbuf,
         DIMC, 1024, 1, 0, 0);
  }

  k_add<<<(MM * DIMC + 255) / 256, 256, 0, stream>>>(hbuf, skipb, MM * DIMC);

  // residual harmonic block, dilations 1,3,5
  const int rdil[3] = {1, 3, 5};
  for (int j = 0; j < 3; ++j) {
    k_tobf16<<<(MM * DIMC + 255) / 256, 256, 0, stream>>>(hbuf, a256, MM * DIMC);
    gemm(a256, q_rw1[j], scales + 10 + j, p.rb1[j], nullptr, nullptr, t1,
         DIMC, DIMC, 3, rdil[j], 1);
    k_snake<<<(MM * DIMC + 255) / 256, 256, 0, stream>>>(t1, p.ra[j], p.rbe[j],
                                                         (float*)nullptr, a256,
                                                         DIMC, MM * DIMC);
    gemm(a256, q_rw2[j], scales + 13 + j, p.rb2[j], nullptr, hbuf, hbuf,
         DIMC, DIMC, 3, 1, 1);
  }

  // heads
  k_rmsnorm<<<MM / 8, 256, 0, stream>>>(hbuf, p.norm_w, a256);
  gemm(a256, q_mag1, scales + 16, p.mag1_b, nullptr, nullptr, t1, DIMC, DIMC, 1, 0, 0);
  k_snake<<<(MM * DIMC + 255) / 256, 256, 0, stream>>>(t1, p.mag_a, p.mag_be,
                                                       (float*)nullptr, a1024,
                                                       DIMC, MM * DIMC);
  gemm(a1024, q_mag2, scales + 17, p.mag2_b, nullptr, nullptr, lm2, FB, DIMC, 1, 0, 0);
  gemm(a256,  q_ph,   scales + 18, p.ph_b,   nullptr, nullptr, phb, FB, DIMC, 1, 0, 0);
  k_tanhpi<<<(MM * FB + 255) / 256, 256, 0, stream>>>(phb, p.cumsum_w, MM * FB);
  k_cumsum<<<(BATCH * FB + 255) / 256, 256, 0, stream>>>(phb);

  // iSTFT: spec pack -> WMMA iDFT GEMM -> overlap-add gather
  k_specpack<<<(MM * KPAD + 255) / 256, 256, 0, stream>>>(lm2, phb, spack);
  gemm(spack, basisb, nullptr, nullptr, nullptr, nullptr, t4, NFFT, KPAD, 1, 0, 0);
  k_istft_out<<<(BATCH * LOUT + 255) / 256, 256, 0, stream>>>(t4, (float*)d_out);
}
